// EGAT_encoder_24876450578636
// MI455X (gfx1250) — matrix operand order repaired and checked by reading, not compiled, not run on hardware
//
#include <hip/hip_runtime.h>
#include <hip/hip_bf16.h>
#include <math.h>

// ---------------------------------------------------------------------------
// EGAT encoder for MI455X (gfx1250, wave32).
// Dense projections: v_wmma_f32_16x16x32_bf16, 64x64 block tile, 8 waves,
// 2 WMMAs per wave per k-step, double-buffered LDS with
// global_load_async_to_lds_b128 (ASYNCcnt) for the A tile.
// Irregular segment ops (edge softmax, aggregation, segment mean) via f32
// global atomics — HBM-bandwidth bound (~10 GB of [E,128] traffic dominates
// the ~0.1 TFLOP of GEMM work at 23.3 TB/s).
//
// Input flattening assumption (setup_inputs() dict insertion order,
// recursing into 'layers' and nested dicts in insertion order):
//   0 node_feat[50000,128] 1 edge_feat[1.6M,64] 2 src[1.6M] 3 dst[1.6M]
//   4+6l .. : layer l: W_ni, W_nj, W_fij, W_node, b_node[128], attn[4,32]
//   22..31  : mlp: W1[64,512], b1, W2[512,256], b2, W3[256,64], b3,
//             gamma[64], beta[64], W4[64,2], b4[2]
// Output: [50000,2] float32.
// ---------------------------------------------------------------------------

typedef __attribute__((ext_vector_type(16))) __bf16 v16bf;
typedef __attribute__((ext_vector_type(8)))  float  v8f;

union Frag16 { v16bf v; unsigned short u[16]; };

__device__ __forceinline__ unsigned short f2bf(float f) {
  unsigned int u = __float_as_uint(f);
  u += 0x7FFFu + ((u >> 16) & 1u);           // round-to-nearest-even
  return (unsigned short)(u >> 16);
}

// --------------------------- conversion ------------------------------------
__global__ void k_f32_to_bf16(const float* __restrict__ in,
                              unsigned short* __restrict__ out, long long n) {
  long long i = (long long)blockIdx.x * blockDim.x + threadIdx.x;
  if (i < n) out[i] = f2bf(in[i]);
}

// --------------------------- WMMA GEMM -------------------------------------
// C[M,N] = act(A[M,K] @ B[K,N] + bias).  A,B bf16 row-major, C f32.
// Block: 256 threads = 8 waves; block tile 64(M) x 64(N); wave (w&3) picks the
// 16-row group, (w>>2) picks the 32-col group -> two 16x16 WMMA tiles/wave.
// K, N multiples of 32/64 respectively. A tile streamed into double-buffered
// LDS with global_load_async_to_lds_b128; B tile transposed via VGPRs.
__global__ __launch_bounds__(256)
void k_gemm_bf16(const unsigned short* __restrict__ A,
                 const unsigned short* __restrict__ B,
                 const float* __restrict__ bias,
                 float* __restrict__ C,
                 int M, int N, int K, int relu)
{
  __shared__ __align__(16) unsigned short lA[2][64][40];   // [buf][m][k] pad
  __shared__ __align__(16) unsigned short lBt[2][64][40];  // [buf][n][k] pad

  const int t    = threadIdx.x;
  const int lane = t & 31;
  const int wave = t >> 5;
  const int m0   = blockIdx.x * 64;
  const int n0   = blockIdx.y * 64;
  const int wm   = (wave & 3) * 16;
  const int wn   = (wave >> 2) * 32;
  const int r    = lane & 15;
  const int hi   = lane >> 4;

  // A staging: thread t moves 16B: row ar, cols ac..ac+7 of the 64x32 tile.
  const int ar   = t >> 2;
  const int ac   = (t & 3) * 8;
  const int arow = m0 + ar;
  // B staging: thread t moves 16B of row k=bk, cols bn..bn+7 (then transpose).
  const int bk   = t >> 3;
  const int bn   = (t & 7) * 8;

  v8f acc0 = {0.f, 0.f, 0.f, 0.f, 0.f, 0.f, 0.f, 0.f};
  v8f acc1 = {0.f, 0.f, 0.f, 0.f, 0.f, 0.f, 0.f, 0.f};

  const int ksteps = K >> 5;

  auto stageA = [&](int ks, int buf) {
    if (arow < M) {
      unsigned lds = (unsigned)(size_t)&lA[buf][ar][ac];
      const unsigned short* g = A + (size_t)arow * K + (ks << 5) + ac;
      asm volatile("global_load_async_to_lds_b128 %0, %1, off"
                   :: "v"(lds), "v"(g) : "memory");
    } else {
      uint4 z = {0u, 0u, 0u, 0u};
      *(uint4*)&lA[buf][ar][ac] = z;
    }
  };
  auto stageB = [&](int ks, int buf) {
    uint4 q = *(const uint4*)(B + (size_t)((ks << 5) + bk) * N + n0 + bn);
    const unsigned short* pv = (const unsigned short*)&q;
#pragma unroll
    for (int j = 0; j < 8; ++j) lBt[buf][bn + j][bk] = pv[j];
  };

  // prologue: stage tile 0
  stageA(0, 0);
  stageB(0, 0);
  asm volatile("s_wait_asynccnt 0x0" ::: "memory");
  __syncthreads();

  int buf = 0;
  for (int ks = 0; ks < ksteps; ++ks) {
    if (ks + 1 < ksteps) {           // stream next tile into the other buffer
      stageA(ks + 1, buf ^ 1);
      stageB(ks + 1, buf ^ 1);
    }
    // A fragment: lane holds row M=r, K = hi*16..+15 (contiguous)
    Frag16 af, bf0, bf1;
#pragma unroll
    for (int i = 0; i < 16; ++i) af.u[i] = lA[buf][wm + r][hi * 16 + i];
    // B fragments: lane holds col N=r (+16), K = hi*16..+15 (contiguous)
#pragma unroll
    for (int i = 0; i < 16; ++i) bf0.u[i] = lBt[buf][wn + r][hi * 16 + i];
#pragma unroll
    for (int i = 0; i < 16; ++i) bf1.u[i] = lBt[buf][wn + 16 + r][hi * 16 + i];

    acc0 = __builtin_amdgcn_wmma_f32_16x16x32_bf16(
        false, af.v, false, bf0.v, (short)0, acc0, false, false);
    acc1 = __builtin_amdgcn_wmma_f32_16x16x32_bf16(
        false, af.v, false, bf1.v, (short)0, acc1, false, false);

    asm volatile("s_wait_asynccnt 0x0" ::: "memory");
    __syncthreads();
    buf ^= 1;
  }

  // Epilogue: lane (r,hi) holds rows hi*8+v, cols r / r+16 of its strip.
  const int col0 = n0 + wn + r;
  float b0 = bias ? bias[col0]      : 0.f;
  float b1 = bias ? bias[col0 + 16] : 0.f;
#pragma unroll
  for (int v = 0; v < 8; ++v) {
    int row = m0 + wm + hi * 8 + v;
    if (row < M) {
      float x0 = acc0[v] + b0;
      float x1 = acc1[v] + b1;
      if (relu) { x0 = fmaxf(x0, 0.f); x1 = fmaxf(x1, 0.f); }
      C[(size_t)row * N + col0]      = x0;
      C[(size_t)row * N + col0 + 16] = x1;
    }
  }
}

// --------------------------- edge attention --------------------------------
// One wave per edge; lane covers 4 of 128 feats (head = lane>>3).
// f_out = leaky_relu(f_fij + f_ni[src] + f_nj[dst])  (in-place over f_fij)
// e[h]  = sum_f f_out * attn[h]  ; atomicMax mapped-f32 segment max by dst.
__global__ __launch_bounds__(256)
void k_edge_attn(float* __restrict__ fbig, const float* __restrict__ fni,
                 const float* __restrict__ fnj, const int* __restrict__ src,
                 const int* __restrict__ dst, const float* __restrict__ attn,
                 float* __restrict__ elog, unsigned int* __restrict__ nmax,
                 int E)
{
  int gid = blockIdx.x * blockDim.x + threadIdx.x;
  int e = gid >> 5;
  if (e >= E) return;
  int lane = threadIdx.x & 31;
  int s = src[e], d = dst[e];
  const float4 fij = *(const float4*)(fbig + (size_t)e * 128 + lane * 4);
  const float4 an  = *(const float4*)(fni  + (size_t)s * 128 + lane * 4);
  const float4 bn  = *(const float4*)(fnj  + (size_t)d * 128 + lane * 4);
  int head = lane >> 3;
  const float* at = attn + head * 32 + (lane & 7) * 4;

  float in[4] = {fij.x + an.x + bn.x, fij.y + an.y + bn.y,
                 fij.z + an.z + bn.z, fij.w + an.w + bn.w};
  float part = 0.f;
  float4 o4;
  float* o = &o4.x;
#pragma unroll
  for (int j = 0; j < 4; ++j) {
    float v = in[j];
    v = v > 0.f ? v : 0.01f * v;
    o[j] = v;
    part += v * at[j];
  }
  *(float4*)(fbig + (size_t)e * 128 + lane * 4) = o4;

  part += __shfl_xor(part, 1, 32);
  part += __shfl_xor(part, 2, 32);
  part += __shfl_xor(part, 4, 32);
  if ((lane & 7) == 0) {
    elog[(size_t)e * 4 + head] = part;
    unsigned int enc = __float_as_uint(part);
    enc = (enc & 0x80000000u) ? ~enc : (enc | 0x80000000u);
    atomicMax(&nmax[(size_t)d * 4 + head], enc);
  }
}

// exp(e - segmax) and segment sum
__global__ void k_softmax(float* __restrict__ elog, const int* __restrict__ dst,
                          const unsigned int* __restrict__ nmax,
                          float* __restrict__ nsum, int E)
{
  long long i = (long long)blockIdx.x * blockDim.x + threadIdx.x;
  if (i >= (long long)E * 4) return;
  int e = (int)(i >> 2), h = (int)(i & 3);
  int d = dst[e];
  unsigned int enc = nmax[(size_t)d * 4 + h];
  unsigned int bits = (enc & 0x80000000u) ? (enc & 0x7FFFFFFFu) : ~enc;
  float m = __uint_as_float(bits);
  float ex = __expf(elog[i] - m);
  elog[i] = ex;
  atomicAdd(&nsum[(size_t)d * 4 + h], ex);
}

// h_out[dst] += h[src] * a   (wave per edge, float4 per lane)
__global__ __launch_bounds__(256)
void k_aggregate(const float* __restrict__ h, const float* __restrict__ elog,
                 const float* __restrict__ nsum, const int* __restrict__ src,
                 const int* __restrict__ dst, float* __restrict__ hout, int E)
{
  int gid = blockIdx.x * blockDim.x + threadIdx.x;
  int e = gid >> 5;
  if (e >= E) return;
  int lane = threadIdx.x & 31;
  int s = src[e], d = dst[e];
  int head = lane >> 3;
  float ax = elog[(size_t)e * 4 + head] / nsum[(size_t)d * 4 + head];
  const float4 hv = *(const float4*)(h + (size_t)s * 128 + lane * 4);
  float* o = hout + (size_t)d * 128 + lane * 4;
  atomicAdd(o + 0, hv.x * ax);
  atomicAdd(o + 1, hv.y * ax);
  atomicAdd(o + 2, hv.z * ax);
  atomicAdd(o + 3, hv.w * ax);
}

// mean over 4 heads: [R,4,32] -> [R,32]
__global__ void k_head_mean(const float* __restrict__ in, float* __restrict__ out,
                            long long R)
{
  long long i = (long long)blockIdx.x * blockDim.x + threadIdx.x;
  if (i >= R * 32) return;
  long long r = i >> 5;
  int f = (int)(i & 31);
  const float* p = in + r * 128 + f;
  out[i] = 0.25f * (p[0] + p[32] + p[64] + p[96]);
}

// segment sum of edge features + degree count (for segment mean)
__global__ void k_seg_accum(const float* __restrict__ w, const int* __restrict__ dst,
                            float* __restrict__ eN, float* __restrict__ cnt, int E)
{
  long long i = (long long)blockIdx.x * blockDim.x + threadIdx.x;
  if (i >= (long long)E * 32) return;
  int e = (int)(i >> 5), f = (int)(i & 31);
  int d = dst[e];
  atomicAdd(&eN[(size_t)d * 32 + f], w[i]);
  if (f == 0) atomicAdd(&cnt[d], 1.0f);
}

// x0 = concat(h, eN/deg)
__global__ void k_concat(const float* __restrict__ hn, const float* __restrict__ eN,
                         const float* __restrict__ cnt, float* __restrict__ x0, int N)
{
  int i = blockIdx.x * blockDim.x + threadIdx.x;
  if (i >= N * 64) return;
  int n = i >> 6, f = i & 63;
  x0[i] = (f < 32) ? hn[n * 32 + f]
                   : eN[n * 32 + (f - 32)] / fmaxf(cnt[n], 1.0f);
}

// batch-norm statistics: one block per column
__global__ __launch_bounds__(256)
void k_bn_stats(const float* __restrict__ x, int M, int C,
                float* __restrict__ mu, float* __restrict__ var)
{
  int c = blockIdx.x;
  __shared__ float ss[256], ss2[256];
  float s = 0.f, s2 = 0.f;
  for (int r = threadIdx.x; r < M; r += 256) {
    float v = x[(size_t)r * C + c];
    s += v; s2 += v * v;
  }
  ss[threadIdx.x] = s; ss2[threadIdx.x] = s2;
  __syncthreads();
  for (int o = 128; o > 0; o >>= 1) {
    if (threadIdx.x < o) { ss[threadIdx.x] += ss[threadIdx.x + o];
                           ss2[threadIdx.x] += ss2[threadIdx.x + o]; }
    __syncthreads();
  }
  if (threadIdx.x == 0) {
    float m = ss[0] / M;
    mu[c] = m;
    var[c] = ss2[0] / M - m * m;
  }
}

// batchnorm + final 64->2 linear
__global__ void k_final(const float* __restrict__ x3, const float* __restrict__ mu,
                        const float* __restrict__ var, const float* __restrict__ gamma,
                        const float* __restrict__ beta, const float* __restrict__ W4,
                        const float* __restrict__ b4, float* __restrict__ out, int N)
{
  int n = blockIdx.x * blockDim.x + threadIdx.x;
  if (n >= N) return;
  float a0 = b4[0], a1 = b4[1];
  for (int f = 0; f < 64; ++f) {
    float xn = (x3[(size_t)n * 64 + f] - mu[f]) * rsqrtf(var[f] + 1e-5f)
               * gamma[f] + beta[f];
    a0 += xn * W4[f * 2 + 0];
    a1 += xn * W4[f * 2 + 1];
  }
  out[n * 2 + 0] = a0;
  out[n * 2 + 1] = a1;
}

// ---------------------------------------------------------------------------
extern "C" void kernel_launch(void* const* d_in, const int* in_sizes, int n_in,
                              void* d_out, int out_size, void* d_ws, size_t ws_size,
                              hipStream_t stream)
{
  (void)n_in; (void)out_size; (void)ws_size;
  const int N = in_sizes[0] / 128;   // 50000 nodes
  const int E = in_sizes[2];         // 1600000 edges

  const float* node_feat = (const float*)d_in[0];
  const float* edge_feat = (const float*)d_in[1];
  const int*   src       = (const int*)d_in[2];
  const int*   dst       = (const int*)d_in[3];
  const int MLP = 4 + 6 * 3;         // 22

  // ---- workspace arena (256B-aligned slots) ----
  char* ws = (char*)d_ws;
  size_t off = 0;
  auto alloc = [&](size_t bytes) -> void* {
    void* p = ws + off;
    off = (off + bytes + 255) & ~(size_t)255;
    return p;
  };
  float*          f_big  = (float*)alloc((size_t)E * 128 * 4);   // f_fij / f_out
  unsigned short* ebf    = (unsigned short*)alloc((size_t)E * 64 * 2);
  float*          edge_w = (float*)alloc((size_t)E * 32 * 4);    // per-layer edge feat
  float*          elog   = (float*)alloc((size_t)E * 4 * 4);
  float*          f_ni   = (float*)alloc((size_t)N * 128 * 4);
  float*          f_nj   = (float*)alloc((size_t)N * 128 * 4);
  float*          h_lin  = (float*)alloc((size_t)N * 128 * 4);
  float*          h_out  = (float*)alloc((size_t)N * 128 * 4);
  unsigned short* nbf    = (unsigned short*)alloc((size_t)N * 128 * 2);
  float*          node_h = (float*)alloc((size_t)N * 32 * 4);
  unsigned int*   nmax   = (unsigned int*)alloc((size_t)N * 4 * 4);
  float*          nsum   = (float*)alloc((size_t)N * 4 * 4);
  float*          eN     = (float*)alloc((size_t)N * 32 * 4);
  float*          cnt    = (float*)alloc((size_t)N * 4);
  unsigned short* wbf_a  = (unsigned short*)alloc(128 * 128 * 2);
  unsigned short* wbf_b  = (unsigned short*)alloc(128 * 128 * 2);
  unsigned short* wbf_c  = (unsigned short*)alloc(128 * 128 * 2);
  unsigned short* wbf_f  = (unsigned short*)alloc(64 * 128 * 2);
  unsigned short* wbf1   = (unsigned short*)alloc(64 * 512 * 2);
  unsigned short* wbf2   = (unsigned short*)alloc(512 * 256 * 2);
  unsigned short* wbf3   = (unsigned short*)alloc(256 * 64 * 2);
  float*          mu     = (float*)alloc(64 * 4);
  float*          var    = (float*)alloc(64 * 4);

  // MLP activations aliased into f_big (free after the GNN layers).
  char* mb = (char*)f_big;
  auto sub = [&](size_t bytes) -> void* {
    void* p = mb;
    mb += (bytes + 255) & ~(size_t)255;
    return p;
  };
  float*          x0  = (float*)sub((size_t)N * 64 * 4);
  unsigned short* x0b = (unsigned short*)sub((size_t)N * 64 * 2);
  float*          x1  = (float*)sub((size_t)N * 512 * 4);
  unsigned short* x1b = (unsigned short*)sub((size_t)N * 512 * 2);
  float*          x2  = (float*)sub((size_t)N * 256 * 4);
  unsigned short* x2b = (unsigned short*)sub((size_t)N * 256 * 2);
  float*          x3  = (float*)sub((size_t)N * 64 * 4);

  auto cdiv = [](long long a, long long b) { return (int)((a + b - 1) / b); };
  auto conv = [&](const float* in, unsigned short* out, long long n) {
    k_f32_to_bf16<<<cdiv(n, 256), 256, 0, stream>>>(in, out, n);
  };
  auto gemm = [&](const unsigned short* A, const unsigned short* B,
                  const float* bias, float* C, int M, int Nc, int K, int relu) {
    dim3 g(cdiv(M, 64), Nc / 64);
    k_gemm_bf16<<<g, 256, 0, stream>>>(A, B, bias, C, M, Nc, K, relu);
  };

  // ---- 3 EGAT layers ----
  const float* nin = node_feat; int nK = 128;
  const float* ein = edge_feat; int eK = 64;
  for (int l = 0; l < 3; ++l) {
    const float* W_ni   = (const float*)d_in[4 + 6 * l + 0];
    const float* W_nj   = (const float*)d_in[4 + 6 * l + 1];
    const float* W_fij  = (const float*)d_in[4 + 6 * l + 2];
    const float* W_node = (const float*)d_in[4 + 6 * l + 3];
    const float* b_node = (const float*)d_in[4 + 6 * l + 4];
    const float* attn   = (const float*)d_in[4 + 6 * l + 5];

    conv(nin, nbf, (long long)N * nK);
    conv(ein, ebf, (long long)E * eK);
    conv(W_ni,   wbf_a, (long long)nK * 128);
    conv(W_nj,   wbf_b, (long long)nK * 128);
    conv(W_node, wbf_c, (long long)nK * 128);
    conv(W_fij,  wbf_f, (long long)eK * 128);

    gemm(nbf, wbf_a, nullptr, f_ni,  N, 128, nK, 0);
    gemm(nbf, wbf_b, nullptr, f_nj,  N, 128, nK, 0);
    gemm(nbf, wbf_c, b_node,  h_lin, N, 128, nK, 0);
    gemm(ebf, wbf_f, nullptr, f_big, E, 128, eK, 0);

    hipMemsetAsync(nmax, 0, (size_t)N * 4 * 4, stream);
    hipMemsetAsync(nsum, 0, (size_t)N * 4 * 4, stream);
    hipMemsetAsync(h_out, 0, (size_t)N * 128 * 4, stream);

    k_edge_attn<<<cdiv((long long)E * 32, 256), 256, 0, stream>>>(
        f_big, f_ni, f_nj, src, dst, attn, elog, nmax, E);
    k_softmax<<<cdiv((long long)E * 4, 256), 256, 0, stream>>>(
        elog, dst, nmax, nsum, E);
    k_aggregate<<<cdiv((long long)E * 32, 256), 256, 0, stream>>>(
        h_lin, elog, nsum, src, dst, h_out, E);
    k_head_mean<<<cdiv((long long)N * 32, 256), 256, 0, stream>>>(h_out, node_h, N);
    k_head_mean<<<cdiv((long long)E * 32, 256), 256, 0, stream>>>(f_big, edge_w, E);

    nin = node_h; nK = 32;
    ein = edge_w; eK = 32;
  }

  // ---- readout: segment mean of edge feats + concat ----
  hipMemsetAsync(eN, 0, (size_t)N * 32 * 4, stream);
  hipMemsetAsync(cnt, 0, (size_t)N * 4, stream);
  k_seg_accum<<<cdiv((long long)E * 32, 256), 256, 0, stream>>>(edge_w, dst, eN, cnt, E);
  k_concat<<<cdiv((long long)N * 64, 256), 256, 0, stream>>>(node_h, eN, cnt, x0, N);

  // ---- MLP ----
  const float* W1 = (const float*)d_in[MLP + 0];
  const float* b1 = (const float*)d_in[MLP + 1];
  const float* W2 = (const float*)d_in[MLP + 2];
  const float* b2 = (const float*)d_in[MLP + 3];
  const float* W3 = (const float*)d_in[MLP + 4];
  const float* b3 = (const float*)d_in[MLP + 5];
  const float* gamma = (const float*)d_in[MLP + 6];
  const float* beta  = (const float*)d_in[MLP + 7];
  const float* W4 = (const float*)d_in[MLP + 8];
  const float* b4 = (const float*)d_in[MLP + 9];

  conv(x0, x0b, (long long)N * 64);
  conv(W1, wbf1, 64 * 512);
  gemm(x0b, wbf1, b1, x1, N, 512, 64, 1);
  conv(x1, x1b, (long long)N * 512);
  conv(W2, wbf2, 512 * 256);
  gemm(x1b, wbf2, b2, x2, N, 256, 512, 1);
  conv(x2, x2b, (long long)N * 256);
  conv(W3, wbf3, 256 * 64);
  gemm(x2b, wbf3, b3, x3, N, 64, 256, 1);

  k_bn_stats<<<64, 256, 0, stream>>>(x3, N, 64, mu, var);
  k_final<<<cdiv(N, 256), 256, 0, stream>>>(x3, mu, var, gamma, beta, W4, b4,
                                            (float*)d_out, N);
}
